// LatentCountingRegisters_88545045775227
// MI455X (gfx1250) — matrix operand order repaired and checked by reading, not compiled, not run on hardware
//
#include <hip/hip_runtime.h>
#include <hip/hip_bf16.h>
#include <math.h>

// ---------------- problem constants ----------------
constexpr int Bn = 64, Hn = 128, Wn = 128, Dn = 128;
constexpr int Cn = 10, En = 128, NFn = 8, NHn = 4;
constexpr int HWn = Hn * Wn;           // 16384
constexpr int HDn = En / NHn;          // 32
constexpr int Sn = 8;                  // segments per batch
constexpr int SEGn = HWn / Sn;         // 2048 keys / segment
constexpr int TKn = 32;                // keys per tile
constexpr int NTn = SEGn / TKn;        // 64 tiles per segment

// ---------------- workspace layout (floats) ----------------
constexpr size_t OFF_CE  = 1024;
constexpr size_t OFF_QH  = OFF_CE + (size_t)Bn * Cn * En;          // 82944
constexpr size_t OFF_M   = OFF_QH + (size_t)Bn * 16 * En;          // 214016
constexpr size_t OFF_L   = OFF_M  + (size_t)Bn * Sn * Cn * NHn;    // 234496
constexpr size_t OFF_CTX = OFF_L  + (size_t)Bn * Sn * Cn * NHn;    // 254976
// total = 910336 floats (~3.65 MB)

typedef __attribute__((ext_vector_type(16))) __bf16 v16bf;
typedef __attribute__((ext_vector_type(8)))  float  v8f;

union FragU  { uint4 u[2]; v16bf v; };     // 32B WMMA operand
union PackU8 { __bf16 h[8]; uint4 u4; };   // 8 bf16 <-> one b128 store

static __device__ __forceinline__ v8f wmma_bf16(v16bf a, v16bf b, v8f c) {
  return __builtin_amdgcn_wmma_f32_16x16x32_bf16(false, a, false, b, (short)0, c, false, false);
}

// Load a 32B operand from LDS as two consecutive b128 reads.
static __device__ __forceinline__ v16bf ld_frag2(const uint4* p) {
  FragU f; f.u[0] = p[0]; f.u[1] = p[1]; return f.v;
}

// 16-bit A-matrix 16x32 fragment from row-major f32 memory (one-time weight/q load).
// Lane layout (contiguous K): element i -> K = kb+i; kb = lane<16?0:16.
static __device__ __forceinline__ v16bf load_afrag_f32(const float* base, int row0, int k0,
                                                       int stride, int lane) {
  const int r  = row0 + (lane & 15);
  const int kb = k0 + ((lane < 16) ? 0 : 16);
  const float* p = base + (size_t)r * stride + kb;
  const float4 f0 = *(const float4*)(p);
  const float4 f1 = *(const float4*)(p + 4);
  const float4 f2 = *(const float4*)(p + 8);
  const float4 f3 = *(const float4*)(p + 12);
  v16bf a;
  a[0] = (__bf16)f0.x;  a[1] = (__bf16)f0.y;  a[2] = (__bf16)f0.z;  a[3] = (__bf16)f0.w;
  a[4] = (__bf16)f1.x;  a[5] = (__bf16)f1.y;  a[6] = (__bf16)f1.z;  a[7] = (__bf16)f1.w;
  a[8] = (__bf16)f2.x;  a[9] = (__bf16)f2.y;  a[10] = (__bf16)f2.z; a[11] = (__bf16)f2.w;
  a[12] = (__bf16)f3.x; a[13] = (__bf16)f3.y; a[14] = (__bf16)f3.z; a[15] = (__bf16)f3.w;
  return a;
}

static __device__ __forceinline__ float gelu_exact(float x) {
  return 0.5f * x * (1.0f + erff(x * 0.70710678118654752f));
}

static __device__ __forceinline__ float block_sum128(float v, volatile float* red4) {
#pragma unroll
  for (int m = 16; m >= 1; m >>= 1) v += __shfl_xor(v, m, 32);
  const int wid = threadIdx.x >> 5;
  if ((threadIdx.x & 31) == 0) red4[wid] = v;
  __syncthreads();
  const float t = red4[0] + red4[1] + red4[2] + red4[3];
  __syncthreads();
  return t;
}

// ---------------- K0: zero counts ----------------
__global__ void k0_zero(int* counts, int n) {
  for (int i = threadIdx.x; i < n; i += blockDim.x) counts[i] = 0;
}

// ---------------- K1: per-batch color histogram ----------------
__global__ void k1_hist(const int* __restrict__ grid, int* __restrict__ counts) {
  const int b = blockIdx.x >> 4;
  const int chunk = blockIdx.x & 15;
  const int base = chunk * (HWn / 16);
  __shared__ int hb[16];
  if (threadIdx.x < 16) hb[threadIdx.x] = 0;
  __syncthreads();
  for (int i = threadIdx.x; i < HWn / 16; i += blockDim.x) {
    int g = grid[(size_t)b * HWn + base + i];
    g = min(max(g, 0), Cn - 1);
    atomicAdd(&hb[g], 1);
  }
  __syncthreads();
  if (threadIdx.x < Cn) atomicAdd(&counts[b * Cn + threadIdx.x], hb[threadIdx.x]);
}

// ---------------- K2: count-embed + q projection ----------------
__global__ void k2_embed(const int* __restrict__ counts,
                         const float* __restrict__ cq,
                         const float* __restrict__ ce_w, const float* __restrict__ ce_b,
                         const float* __restrict__ ce_ln_g, const float* __restrict__ ce_ln_b,
                         const float* __restrict__ in_proj_w, const float* __restrict__ in_proj_b,
                         const float* __restrict__ freqs,
                         float* __restrict__ ce_ws, float* __restrict__ qh_ws) {
  const int b = blockIdx.x;
  const int e = threadIdx.x;  // 128
  __shared__ float sh_q[En];
  __shared__ float red4[4];
  const float PI = 3.14159265358979323846f;

  for (int c = 0; c < Cn; ++c) {
    const float cn = (float)counts[b * Cn + c] / (float)HWn;
    float acc = ce_b[e] + cn * ce_w[e];  // row 0
#pragma unroll
    for (int j = 0; j < NFn; ++j) {
      const float ang = cn * freqs[j] * PI;
      acc += sinf(ang) * ce_w[(1 + j) * En + e];
      acc += cosf(ang) * ce_w[(1 + NFn + j) * En + e];
    }
    const float g = gelu_exact(acc);
    const float mean = block_sum128(g, red4) * (1.0f / En);
    const float d = g - mean;
    const float var = block_sum128(d * d, red4) * (1.0f / En);
    const float cev = d * rsqrtf(var + 1e-5f) * ce_ln_g[e] + ce_ln_b[e];
    ce_ws[((size_t)b * Cn + c) * En + e] = cev;
    sh_q[e] = cq[c * En + e] + cev;
    __syncthreads();
    float qa = in_proj_b[e];
    for (int e2 = 0; e2 < En; ++e2) qa += sh_q[e2] * in_proj_w[(size_t)e * En + e2];
    qh_ws[((size_t)b * 16 + c) * En + e] = qa * 0.17677669529663687f;  // 1/sqrt(HD)
    __syncthreads();
  }
  for (int c = Cn; c < 16; ++c) qh_ws[((size_t)b * 16 + c) * En + e] = 0.0f;
}

// ---------------- K3: fused K/V projection + masked flash attention ----------------
__global__ void __launch_bounds__(128)
k3_attn(const int* __restrict__ grid, const float* __restrict__ features,
        const float* __restrict__ in_proj_w, const float* __restrict__ in_proj_b,
        const float* __restrict__ qh_ws,
        float* __restrict__ m_ws, float* __restrict__ l_ws, float* __restrict__ ctx_ws) {
  const int b   = blockIdx.x / Sn;
  const int seg = blockIdx.x % Sn;
  const int tid = threadIdx.x;
  const int wave = tid >> 5;          // wave == head
  const int lane = tid & 31;
  const int colk = lane & 15;
  const int hi8  = (lane < 16) ? 0 : 8;
  const int hi16 = (lane < 16) ? 0 : 16;
  const int head = wave;

  // All tiles stored so the WMMA K-direction is CONTIGUOUS -> b128 fragment loads.
  __shared__ uint4 featT[TKn][17];        // [key][d/8]  : d contiguous (272B rows)
  __shared__ uint4 kT[NHn][TKn][5];       // [head][key][hd/8] : hd contiguous
  __shared__ uint4 vT[NHn][HDn][5];       // [head][hd][key/8] : key contiguous
  __shared__ uint4 pT[NHn][16][5];        // [head][c][key/8]  : key contiguous

  // ---- register-resident bf16 weight fragments (reused over 64 tiles) ----
  const float* Wk = in_proj_w + (size_t)En * En;
  const float* Wv = in_proj_w + (size_t)2 * En * En;
  v16bf ka[2][4], va[2][4];
#pragma unroll
  for (int m2 = 0; m2 < 2; ++m2)
#pragma unroll
    for (int k4 = 0; k4 < 4; ++k4) {
      ka[m2][k4] = load_afrag_f32(Wk, head * HDn + m2 * 16, k4 * 32, En, lane);
      va[m2][k4] = load_afrag_f32(Wv, head * HDn + m2 * 16, k4 * 32, En, lane);
    }
  const v16bf qa = load_afrag_f32(qh_ws + (size_t)b * 16 * En, 0, head * HDn, En, lane);

  // K-bias folded into a per-row score bias: sb[c] = q_c . bk  (q already /sqrt(HD))
  float sb_r[8];
  {
    float sb = 0.f;
    const float* qrow = qh_ws + ((size_t)b * 16 + colk) * En + head * HDn;
    const float* bk = in_proj_b + En + head * HDn;
#pragma unroll
    for (int hd = 0; hd < HDn; ++hd) sb += qrow[hd] * bk[hd];
#pragma unroll
    for (int v = 0; v < 8; ++v) sb_r[v] = __shfl(sb, v + hi8, 32);
  }

  const v8f z8 = {0.f, 0.f, 0.f, 0.f, 0.f, 0.f, 0.f, 0.f};
  v8f ctx0 = z8, ctx1 = z8;
  float mrow[8], lrow[8];
#pragma unroll
  for (int v = 0; v < 8; ++v) { mrow[v] = -1e30f; lrow[v] = 0.0f; }

  const float* fbase = features + (size_t)b * Dn * HWn;
  const int dq = wave * 32;                  // this wave stages d in [dq, dq+32)
  __bf16* vT_h = (__bf16*)&vT[head][0][0];   // row stride 40 bf16
  __bf16* pT_h = (__bf16*)&pT[head][0][0];   // row stride 40 bf16

  // ---- software-pipelined staging: tile t+1 loaded/packed in regs during tile t compute ----
  uint4 stage[4];
  {
    const float* src = fbase + seg * SEGn + lane;   // tile 0
#pragma unroll
    for (int j = 0; j < 4; ++j) {
      PackU8 pu;
#pragma unroll
      for (int u = 0; u < 8; ++u) pu.h[u] = (__bf16)src[(size_t)(dq + j * 8 + u) * HWn];
      stage[j] = pu.u4;
    }
  }

  for (int t = 0; t < NTn; ++t) {
    const int n0 = seg * SEGn + t * TKn;

    // featT free here (kernel start or end-of-previous-tile barrier)
#pragma unroll
    for (int j = 0; j < 4; ++j) featT[lane][wave * 4 + j] = stage[j];
    __syncthreads();

    // issue next tile's global loads early; they retire behind this tile's compute
    if (t + 1 < NTn) {
      const float* src = fbase + n0 + TKn + lane;
#pragma unroll
      for (int j = 0; j < 4; ++j) {
        PackU8 pu;
#pragma unroll
        for (int u = 0; u < 8; ++u) pu.h[u] = (__bf16)src[(size_t)(dq + j * 8 + u) * HWn];
        stage[j] = pu.u4;
        __builtin_prefetch(src + TKn + (size_t)(dq + j * 8) * HWn, 0, 0);  // tile t+2 -> L2
      }
    }

    // ---- K/V projection for this head: (32e x 128d) @ (128d x 32n) ----
#pragma unroll
    for (int n2 = 0; n2 < 2; ++n2) {
      v8f kacc[2], vacc[2];
      kacc[0] = z8; kacc[1] = z8; vacc[0] = z8; vacc[1] = z8;
#pragma unroll
      for (int k4 = 0; k4 < 4; ++k4) {
        const v16bf bf = ld_frag2(&featT[n2 * 16 + colk][k4 * 4 + (hi16 >> 3)]);
#pragma unroll
        for (int m2 = 0; m2 < 2; ++m2) {
          kacc[m2] = wmma_bf16(ka[m2][k4], bf, kacc[m2]);
          vacc[m2] = wmma_bf16(va[m2][k4], bf, vacc[m2]);
        }
      }
      // K tile: rows (vgpr dir) are hd-contiguous -> packed b128 stores
#pragma unroll
      for (int m2 = 0; m2 < 2; ++m2) {
        PackU8 pk;
#pragma unroll
        for (int v = 0; v < 8; ++v) pk.h[v] = (__bf16)kacc[m2][v];
        kT[head][n2 * 16 + colk][m2 * 2 + (hi8 >> 3)] = pk.u4;
        // V tile needs [hd][key] for the ctx B-fragment -> scatter b16 stores
#pragma unroll
        for (int v = 0; v < 8; ++v)
          vT_h[(m2 * 16 + v + hi8) * 40 + n2 * 16 + colk] = (__bf16)vacc[m2][v];
      }
    }
    asm volatile("s_wait_dscnt 0" ::: "memory");  // intra-wave DS write->read ordering

    // ---- scores: (16c x 32hd) @ (32hd x 16key) per n-tile ----
    v8f sacc[2];
#pragma unroll
    for (int n2 = 0; n2 < 2; ++n2)
      sacc[n2] = wmma_bf16(qa, ld_frag2(&kT[head][n2 * 16 + colk][hi16 >> 3]), z8);

    // ---- masking + online softmax ----
    const int g0 = grid[(size_t)b * HWn + n0 + colk];
    const int g1 = grid[(size_t)b * HWn + n0 + 16 + colk];
#pragma unroll
    for (int v = 0; v < 8; ++v) {
      const int c = v + hi8;
      float s0 = (g0 == c) ? (sacc[0][v] + sb_r[v]) : -1e30f;
      float s1 = (g1 == c) ? (sacc[1][v] + sb_r[v]) : -1e30f;
      float mx = fmaxf(s0, s1);
#pragma unroll
      for (int mk = 8; mk >= 1; mk >>= 1) mx = fmaxf(mx, __shfl_xor(mx, mk, 32));
      const float mnew = fmaxf(mrow[v], mx);
      const float scale = __expf(mrow[v] - mnew);
      const float p0 = __expf(s0 - mnew);
      const float p1 = __expf(s1 - mnew);
      float rs = p0 + p1;
#pragma unroll
      for (int mk = 8; mk >= 1; mk >>= 1) rs += __shfl_xor(rs, mk, 32);
      lrow[v] = lrow[v] * scale + rs;
      mrow[v] = mnew;
      ctx0[v] *= scale;
      ctx1[v] *= scale;
      pT_h[c * 40 + colk]      = (__bf16)p0;
      pT_h[c * 40 + 16 + colk] = (__bf16)p1;
    }
    asm volatile("s_wait_dscnt 0" ::: "memory");

    // ---- ctx += P (16c x 32key) @ V (32key x 32hd) ----
    v16bf pa;
    {
      FragU f;
      f.u[0] = pT[head][colk][hi16 >> 3];
      f.u[1] = pT[head][colk][(hi16 >> 3) + 1];
      pa = f.v;
    }
    ctx0 = wmma_bf16(pa, ld_frag2(&vT[head][colk][hi16 >> 3]), ctx0);
    ctx1 = wmma_bf16(pa, ld_frag2(&vT[head][16 + colk][hi16 >> 3]), ctx1);

    __syncthreads();  // all waves done with featT before next tile's store
  }

  // ---- write per-segment partials (unnormalized ctx, m, l) ----
#pragma unroll
  for (int v = 0; v < 8; ++v) {
    const int c = v + hi8;
    if (c < Cn) {
      const size_t base = ((((size_t)b * Sn + seg) * Cn + c) * NHn + head);
      ctx_ws[base * HDn + colk]      = ctx0[v];
      ctx_ws[base * HDn + 16 + colk] = ctx1[v];
      if (colk == 0) { m_ws[base] = mrow[v]; l_ws[base] = lrow[v]; }
    }
  }
}

// ---------------- K4: segment merge + V-bias + out_proj + final MLP + LN ----------------
__global__ void k4_out(const float* __restrict__ ctx_ws, const float* __restrict__ m_ws,
                       const float* __restrict__ l_ws, const int* __restrict__ counts,
                       const float* __restrict__ ce_ws, const float* __restrict__ in_proj_b,
                       const float* __restrict__ out_proj_w, const float* __restrict__ out_proj_b,
                       const float* __restrict__ op_w, const float* __restrict__ op_b,
                       const float* __restrict__ op_ln_g, const float* __restrict__ op_ln_b,
                       float* __restrict__ out) {
  const int b = blockIdx.x;
  const int e = threadIdx.x;  // 128
  __shared__ float ctxrow[En];
  __shared__ float comb[2 * En];
  __shared__ float red4[4];

  for (int c = 0; c < Cn; ++c) {
    const int h = e >> 5, hd = e & 31;
    float M = -1e30f;
#pragma unroll
    for (int s = 0; s < Sn; ++s)
      M = fmaxf(M, m_ws[(((size_t)b * Sn + s) * Cn + c) * NHn + h]);
    float L = 0.f, acc = 0.f;
    if (M > -1e29f) {
#pragma unroll
      for (int s = 0; s < Sn; ++s) {
        const size_t base = (((size_t)b * Sn + s) * Cn + c) * NHn + h;
        const float ms = m_ws[base];
        if (ms > -1e29f) {
          const float w = __expf(ms - M);
          L += w * l_ws[base];
          acc += w * ctx_ws[base * HDn + hd];
        }
      }
    }
    // V-bias added here (softmax weights sum to 1)
    ctxrow[e] = ((L > 0.f) ? acc / L : 0.f) + in_proj_b[2 * En + e];
    __syncthreads();

    float att = out_proj_b[e];
    for (int e2 = 0; e2 < En; ++e2) att += ctxrow[e2] * out_proj_w[(size_t)e * En + e2];
    att *= (counts[b * Cn + c] > 0) ? 1.0f : 0.0f;

    comb[e] = ce_ws[((size_t)b * Cn + c) * En + e];
    comb[En + e] = att;
    __syncthreads();

    float g = op_b[e];
    for (int k = 0; k < 2 * En; ++k) g += comb[k] * op_w[(size_t)k * En + e];
    g = gelu_exact(g);

    const float mean = block_sum128(g, red4) * (1.0f / En);
    const float d = g - mean;
    const float var = block_sum128(d * d, red4) * (1.0f / En);
    out[((size_t)b * Cn + c) * En + e] = d * rsqrtf(var + 1e-5f) * op_ln_g[e] + op_ln_b[e];
    __syncthreads();
  }
}

// ---------------- launcher ----------------
extern "C" void kernel_launch(void* const* d_in, const int* in_sizes, int n_in,
                              void* d_out, int out_size, void* d_ws, size_t ws_size,
                              hipStream_t stream) {
  (void)in_sizes; (void)n_in; (void)out_size; (void)ws_size;
  const int*   grid_in    = (const int*)d_in[0];
  const float* features   = (const float*)d_in[1];
  const float* cq         = (const float*)d_in[2];
  const float* ce_w       = (const float*)d_in[3];
  const float* ce_b       = (const float*)d_in[4];
  const float* ce_ln_g    = (const float*)d_in[5];
  const float* ce_ln_b    = (const float*)d_in[6];
  const float* in_proj_w  = (const float*)d_in[7];
  const float* in_proj_b  = (const float*)d_in[8];
  const float* out_proj_w = (const float*)d_in[9];
  const float* out_proj_b = (const float*)d_in[10];
  const float* op_w       = (const float*)d_in[11];
  const float* op_b       = (const float*)d_in[12];
  const float* op_ln_g    = (const float*)d_in[13];
  const float* op_ln_b    = (const float*)d_in[14];
  const float* freqs      = (const float*)d_in[15];

  float* ws = (float*)d_ws;
  int*   counts = (int*)d_ws;
  float* ce_ws  = ws + OFF_CE;
  float* qh_ws  = ws + OFF_QH;
  float* m_ws   = ws + OFF_M;
  float* l_ws   = ws + OFF_L;
  float* ctx_ws = ws + OFF_CTX;

  k0_zero<<<1, 256, 0, stream>>>(counts, Bn * Cn);
  k1_hist<<<Bn * 16, 256, 0, stream>>>(grid_in, counts);
  k2_embed<<<Bn, 128, 0, stream>>>(counts, cq, ce_w, ce_b, ce_ln_g, ce_ln_b,
                                   in_proj_w, in_proj_b, freqs, ce_ws, qh_ws);
  k3_attn<<<Bn * Sn, 128, 0, stream>>>(grid_in, features, in_proj_w, in_proj_b,
                                       qh_ws, m_ws, l_ws, ctx_ws);
  k4_out<<<Bn, 128, 0, stream>>>(ctx_ws, m_ws, l_ws, counts, ce_ws, in_proj_b,
                                 out_proj_w, out_proj_b, op_w, op_b,
                                 op_ln_g, op_ln_b, (float*)d_out);
}
